// Attention_80384607912675
// MI455X (gfx1250) — matrix operand order repaired and checked by reading, not compiled, not run on hardware
//
#include <hip/hip_runtime.h>
#include <hip/hip_bf16.h>

// ---------------------------------------------------------------------------
// MI455X (gfx1250) fused multi-head attention forward:
//   out = softmax((xWq+bq)(xWk+bk)^T / sqrt(HD)) (xWv+bv) Wo + bo
// bf16 WMMA everywhere; flash-style tiling (scores never touch HBM);
// CDNA5 async global->LDS staging for the non-transposed tiles.
// ---------------------------------------------------------------------------

typedef __attribute__((ext_vector_type(16))) __bf16 v16bf;
typedef __attribute__((ext_vector_type(8)))  float  v8f;

union U32BF2 { unsigned int u; __bf16 h[2]; };
union Frag   { unsigned int u[8]; v16bf v; };

__device__ __forceinline__ v8f wmma_bf16(const Frag& a, const Frag& b, v8f c) {
  // 8 args: (neg_a, A, neg_b, B, c_mod, C, reuse_a, reuse_b)
  return __builtin_amdgcn_wmma_f32_16x16x32_bf16(false, a.v, false, b.v,
                                                 (short)0, c, false, false);
}

// LDS byte offset = low 32 bits of the generic (flat) shared-aperture address.
__device__ __forceinline__ unsigned lds_off(const void* p) {
  return (unsigned)(unsigned long long)p;
}
// CDNA5 async DMA: 16B per lane, global -> LDS, tracked by ASYNCcnt.
__device__ __forceinline__ void async_b128(unsigned dst_lds, const void* src) {
  asm volatile("global_load_async_to_lds_b128 %0, %1, off"
               :: "v"(dst_lds), "v"(src) : "memory");
}
__device__ __forceinline__ void wait_async0() {
  asm volatile("s_wait_asynccnt 0x0" ::: "memory");
}

// ---------------------------------------------------------------------------
// f32 -> bf16 conversion
// ---------------------------------------------------------------------------
__global__ void f32_to_bf16(const float* __restrict__ in,
                            __bf16* __restrict__ out, int n) {
  int i = blockIdx.x * 256 + threadIdx.x;
  if (i < n) out[i] = (__bf16)in[i];
}

// ---------------------------------------------------------------------------
// Tiled bf16 GEMM with WMMA:  C[M,N] = A[M,K] * W[K,N] + bias
//   128 threads = 4 waves; tile 64x64; K-step 32.
//   A tile staged by GLOBAL_LOAD_ASYNC_TO_LDS_B128 (row-major, no transform);
//   W tile transposed manually into [n][k] so B-fragments are pair-contiguous.
// ---------------------------------------------------------------------------
template <int OUT_F32>
__global__ __launch_bounds__(128)
void gemm_bf16_wmma(const __bf16* __restrict__ A, const __bf16* __restrict__ W,
                    const float* __restrict__ bias,
                    __bf16* __restrict__ outb, float* __restrict__ outf,
                    int M, int N, int K)
{
  // A tile: 64 rows x 32 bf16, row stride 20 uints (80B, 16B aligned for B128).
  __shared__ __align__(16) unsigned int As_u[64 * 20];
  // W tile transposed [n][k]: row stride 18 uints (pair loads stay 4B aligned).
  __shared__ unsigned int Bs_u[64 * 18];
  __bf16* Bs_b = (__bf16*)Bs_u;                 // row stride 36 bf16

  const int tid  = threadIdx.x;
  const int wv_  = tid >> 5;
  const int lane = tid & 31;
  const int hi   = lane >> 4;
  const int l16  = lane & 15;
  const int m0   = blockIdx.x * 64;
  const int n0   = blockIdx.y * 64;

  const unsigned int* Wg = (const unsigned int*)W;
  const int NU = N >> 1;
  const unsigned as0 = lds_off(As_u);

  v8f acc[4] = {};

  for (int k0 = 0; k0 < K; k0 += 32) {
    // ---- async-stage A tile: 64 rows x 4 x 16B chunks, 2 chunks per thread
#pragma unroll
    for (int i = 0; i < 2; ++i) {
      int s = tid + i * 128;                    // 0..255
      int r = s >> 2, c = s & 3;
      const __bf16* src = A + (size_t)(m0 + r) * K + k0 + c * 8;
      async_b128(as0 + r * 80 + c * 16, src);
    }
    // prefetch next A tile into cache hierarchy (global_prefetch_b8)
    if (k0 + 32 < K)
      __builtin_prefetch(A + (size_t)(m0 + (tid >> 1)) * K + k0 + 32, 0, 1);

    // ---- stage W tile transposed: W[k0+kk][n0+2*cn .. +1] -> Bs[n][kk]
#pragma unroll
    for (int i = 0; i < 8; ++i) {
      int s = tid + i * 128;                    // 32 kk rows x 32 uint cols
      int kk = s >> 5, cn = s & 31;
      U32BF2 t; t.u = Wg[(size_t)(k0 + kk) * NU + (n0 >> 1) + cn];
      Bs_b[(cn * 2 + 0) * 36 + kk] = t.h[0];
      Bs_b[(cn * 2 + 1) * 36 + kk] = t.h[1];
    }
    wait_async0();                              // ASYNCcnt -> 0 before barrier
    __syncthreads();

    // ---- A fragment (16x32 bf16): uint idx (p&3)+hi*4+(p>>2)*8, row stride 20
    Frag a;
#pragma unroll
    for (int p = 0; p < 8; ++p)
      a.u[p] = As_u[(wv_ * 16 + l16) * 20 + (p & 3) + hi * 4 + ((p >> 2) << 3)];

#pragma unroll
    for (int nt = 0; nt < 4; ++nt) {
      // B fragment (32x16): contraction pair idx (p&3)+hi*4+(p>>2)*8, lane = N
      Frag bfr;
#pragma unroll
      for (int p = 0; p < 8; ++p)
        bfr.u[p] = Bs_u[(nt * 16 + l16) * 18 + (p & 3) + hi * 4 + ((p >> 2) << 3)];
      acc[nt] = wmma_bf16(a, bfr, acc[nt]);
    }
    __syncthreads();
  }

  // ---- epilogue: C layout VGPR r -> row r + 8*hi, col = lane&15
#pragma unroll
  for (int nt = 0; nt < 4; ++nt) {
    int col = n0 + nt * 16 + l16;
    float bv = bias[col];
#pragma unroll
    for (int r = 0; r < 8; ++r) {
      int row = m0 + wv_ * 16 + r + hi * 8;
      float v = acc[nt][r] + bv;
      if (OUT_F32) outf[(size_t)row * N + col] = v;
      else         outb[(size_t)row * N + col] = (__bf16)v;
    }
  }
}

// ---------------------------------------------------------------------------
// Flash attention: one workgroup = one (b,h) x 64-query block.
//   K block async-staged row-major; V block transposed manually;
//   S = Q K^T (WMMA) -> online softmax in LDS -> O = O*fac + P V (WMMA).
// Q/K/V/O stored [B*S, D] with head-major columns (col = h*64 + d).
// ---------------------------------------------------------------------------
__global__ __launch_bounds__(128)
void attn_fa_wmma(const __bf16* __restrict__ Q, const __bf16* __restrict__ K,
                  const __bf16* __restrict__ V, __bf16* __restrict__ O,
                  int Sx, int Hx, int Dx, float scale)
{
  // K block [key][d]: row stride 36 uints (144B, 16B aligned for async B128)
  __shared__ __align__(16) unsigned int Kt_u[64 * 36];
  __shared__ unsigned int Vt_u[64 * 33];   // V transposed [d][key]
  __shared__ unsigned int Pt_u[64 * 33];   // P (softmax probs) [q][key] bf16
  __shared__ float Sc[64][65];             // raw scores f32
  __shared__ float m_s[64], l_s[64], fac_s[64];
  __shared__ float redm[64][2], reds[64][2];
  __bf16* Vt_b = (__bf16*)Vt_u;            // row stride 66 bf16
  __bf16* Pt_b = (__bf16*)Pt_u;

  const int tid  = threadIdx.x;
  const int wv_  = tid >> 5;
  const int lane = tid & 31;
  const int hi   = lane >> 4;
  const int l16  = lane & 15;
  const int bh   = blockIdx.y;
  const int b    = bh / Hx, h = bh % Hx;
  const int q0   = blockIdx.x * 64;

  const unsigned int* Qg = (const unsigned int*)Q;
  const unsigned int* Vg = (const unsigned int*)V;
  const size_t rowU  = (size_t)(Dx >> 1);   // uints per row
  const int    cbase = (h * 64) >> 1;       // uint column offset of this head
  const unsigned kt0 = lds_off(Kt_u);

  // ---- Q fragments for this wave's 16 rows, resident for the whole loop
  Frag qf[2];
  {
    size_t qrow = (size_t)(b * Sx + q0 + wv_ * 16 + l16) * rowU + cbase;
#pragma unroll
    for (int ks = 0; ks < 2; ++ks)
#pragma unroll
      for (int p = 0; p < 8; ++p)
        qf[ks].u[p] = Qg[qrow + (p & 3) + hi * 4 + ((p >> 2) << 3) + ks * 16];
  }

  if (tid < 64) { m_s[tid] = -1e30f; l_s[tid] = 0.0f; }
  v8f Oacc[4] = {};
  __syncthreads();

  for (int kb = 0; kb < Sx; kb += 64) {
    // ---- async-stage K block: 64 rows x 8 x 16B chunks, 4 per thread
#pragma unroll
    for (int i = 0; i < 4; ++i) {
      int s = tid + i * 128;                    // 0..511
      int r = s >> 3, c = s & 7;
      const __bf16* src = K + (size_t)(b * Sx + kb + r) * Dx + h * 64 + c * 8;
      async_b128(kt0 + r * 144 + c * 16, src);
    }
    // ---- stage V block transposed: V[key][d] -> Vt[d][key]
#pragma unroll
    for (int i = 0; i < 16; ++i) {
      int s = tid + i * 128;
      int r = s >> 5, cu = s & 31;
      U32BF2 t; t.u = Vg[(size_t)(b * Sx + kb + r) * rowU + cbase + cu];
      Vt_b[(cu * 2 + 0) * 66 + r] = t.h[0];
      Vt_b[(cu * 2 + 1) * 66 + r] = t.h[1];
    }
    wait_async0();
    __syncthreads();

    // ---- scores: S = Q K^T  (contraction = d, B-frag N = key)
#pragma unroll
    for (int nt = 0; nt < 4; ++nt) {
      v8f s8 = {};
#pragma unroll
      for (int ks = 0; ks < 2; ++ks) {
        Frag bfr;
#pragma unroll
        for (int p = 0; p < 8; ++p)
          bfr.u[p] = Kt_u[(nt * 16 + l16) * 36 + (p & 3) + hi * 4 + ((p >> 2) << 3) + ks * 16];
        s8 = wmma_bf16(qf[ks], bfr, s8);
      }
#pragma unroll
      for (int r = 0; r < 8; ++r)
        Sc[wv_ * 16 + r + hi * 8][nt * 16 + l16] = s8[r] * scale;
    }
    __syncthreads();

    // ---- online softmax: 2 threads per query row, 32 keys each
    {
      int row = tid >> 1, half = tid & 1;
      int c0 = half * 32;
      float lm = -1e30f;
#pragma unroll
      for (int c = 0; c < 32; ++c) lm = fmaxf(lm, Sc[row][c0 + c]);
      redm[row][half] = lm;
      __syncthreads();
      float mo = m_s[row];
      float mn = fmaxf(mo, fmaxf(redm[row][0], redm[row][1]));
      float ls = 0.0f;
#pragma unroll
      for (int c = 0; c < 32; ++c) {
        float p = __expf(Sc[row][c0 + c] - mn);
        ls += p;
        Pt_b[row * 66 + c0 + c] = (__bf16)p;
      }
      reds[row][half] = ls;
      __syncthreads();
      if (half == 0) {
        float fac = __expf(mo - mn);       // 0 on first block (mo = -1e30)
        m_s[row]   = mn;
        l_s[row]   = l_s[row] * fac + reds[row][0] + reds[row][1];
        fac_s[row] = fac;
      }
    }
    __syncthreads();

    // ---- O = O*fac + P V   (A-frag = P from LDS, B-frag = V^T, N = d)
    Frag pf[2];
#pragma unroll
    for (int ks = 0; ks < 2; ++ks)
#pragma unroll
      for (int p = 0; p < 8; ++p)
        pf[ks].u[p] =
            Pt_u[(wv_ * 16 + l16) * 33 + (p & 3) + hi * 4 + ((p >> 2) << 3) + ks * 16];
#pragma unroll
    for (int nt = 0; nt < 4; ++nt) {
#pragma unroll
      for (int r = 0; r < 8; ++r)
        Oacc[nt][r] *= fac_s[wv_ * 16 + r + hi * 8];
#pragma unroll
      for (int ks = 0; ks < 2; ++ks) {
        Frag bfr;
#pragma unroll
        for (int p = 0; p < 8; ++p)
          bfr.u[p] = Vt_u[(nt * 16 + l16) * 33 + (p & 3) + hi * 4 + ((p >> 2) << 3) + ks * 16];
        Oacc[nt] = wmma_bf16(pf[ks], bfr, Oacc[nt]);
      }
    }
    __syncthreads();
  }

  // ---- normalize and write bf16 attention output
#pragma unroll
  for (int nt = 0; nt < 4; ++nt) {
#pragma unroll
    for (int r = 0; r < 8; ++r) {
      int rl = wv_ * 16 + r + hi * 8;
      float v = Oacc[nt][r] / l_s[rl];
      size_t idx = (size_t)(b * Sx + q0 + rl) * Dx + h * 64 + nt * 16 + l16;
      O[idx] = (__bf16)v;
    }
  }
}

// ---------------------------------------------------------------------------
// Host-side orchestration (graph-capture safe: only kernel launches on stream)
// ---------------------------------------------------------------------------
extern "C" void kernel_launch(void* const* d_in, const int* in_sizes, int n_in,
                              void* d_out, int out_size, void* d_ws, size_t ws_size,
                              hipStream_t stream)
{
  const int B = 2, S = 2048, D = 1024, H = 16;
  const int M = B * S;                       // 4096 rows

  const float* x  = (const float*)d_in[0];
  const float* wq = (const float*)d_in[1];
  const float* bq = (const float*)d_in[2];
  const float* wk = (const float*)d_in[3];
  const float* bk = (const float*)d_in[4];
  const float* wv = (const float*)d_in[5];
  const float* bv = (const float*)d_in[6];
  const float* wo = (const float*)d_in[7];
  const float* bo = (const float*)d_in[8];

  // workspace layout (bf16): x | wq wk wv wo | Q K V | attn_out  (~48 MB)
  char* w = (char*)d_ws;
  __bf16* xb  = (__bf16*)w; w += (size_t)M * D * 2;
  __bf16* wqb = (__bf16*)w; w += (size_t)D * D * 2;
  __bf16* wkb = (__bf16*)w; w += (size_t)D * D * 2;
  __bf16* wvb = (__bf16*)w; w += (size_t)D * D * 2;
  __bf16* wob = (__bf16*)w; w += (size_t)D * D * 2;
  __bf16* Qb  = (__bf16*)w; w += (size_t)M * D * 2;
  __bf16* Kb  = (__bf16*)w; w += (size_t)M * D * 2;
  __bf16* Vb  = (__bf16*)w; w += (size_t)M * D * 2;
  __bf16* Ob  = (__bf16*)w; w += (size_t)M * D * 2;

  const int nX = M * D, nW = D * D;
  f32_to_bf16<<<(nX + 255) / 256, 256, 0, stream>>>(x,  xb,  nX);
  f32_to_bf16<<<(nW + 255) / 256, 256, 0, stream>>>(wq, wqb, nW);
  f32_to_bf16<<<(nW + 255) / 256, 256, 0, stream>>>(wk, wkb, nW);
  f32_to_bf16<<<(nW + 255) / 256, 256, 0, stream>>>(wv, wvb, nW);
  f32_to_bf16<<<(nW + 255) / 256, 256, 0, stream>>>(wo, wob, nW);

  dim3 ggrid(M / 64, D / 64);                // 64 x 16 workgroups
  gemm_bf16_wmma<0><<<ggrid, 128, 0, stream>>>(xb, wqb, bq, Qb, nullptr, M, D, D);
  gemm_bf16_wmma<0><<<ggrid, 128, 0, stream>>>(xb, wkb, bk, Kb, nullptr, M, D, D);
  gemm_bf16_wmma<0><<<ggrid, 128, 0, stream>>>(xb, wvb, bv, Vb, nullptr, M, D, D);

  dim3 agrid(S / 64, B * H);                 // 32 x 32 workgroups
  attn_fa_wmma<<<agrid, 128, 0, stream>>>(Qb, Kb, Vb, Ob, S, H, D, 0.125f);

  gemm_bf16_wmma<1><<<ggrid, 128, 0, stream>>>(Ob, wob, bo, nullptr,
                                               (float*)d_out, M, D, D);
}
